// AttentionHead_63771674411557
// MI455X (gfx1250) — compile-verified, matrix operand order checked by reading, not run on hardware
//
#include <hip/hip_runtime.h>
#include <hip/hip_bf16.h>

typedef __attribute__((ext_vector_type(16))) _Float16 v16h;
typedef __attribute__((ext_vector_type(8)))  _Float16 v8h;
typedef __attribute__((ext_vector_type(8)))  float    v8f;

#define S_LEN 4096
#define E_DIM 1024
#define H_DIM 64

__device__ __forceinline__ v8f zero8() {
  v8f z;
#pragma unroll
  for (int i = 0; i < 8; ++i) z[i] = 0.f;
  return z;
}

// A/B fragment (16x32 A or 32x16 B, f16) from a row-major f16 source row.
// Lane layout (ISA 7.12.2, 16-bit 16x32): lane holds K = {c..c+7} U {c+16..c+23},
// c = 8*(lane>=16); row (A: M, B: N) = lane%16 (caller bakes row into `row`).
__device__ __forceinline__ v16h frag_from_f16(const _Float16* __restrict__ row,
                                              int kbase, int chalf) {
  union { v16h v; struct { v8h lo, hi; } s; } u;
  u.s.lo = *(const v8h*)(row + kbase + chalf);
  u.s.hi = *(const v8h*)(row + kbase + chalf + 16);
  return u.v;
}

// ---- CDNA5 async global->LDS copy (ASYNCcnt-tracked; ISA 08_async_tensor §4)
__device__ __forceinline__ void async_load_b128(unsigned lds_off, const void* g) {
  asm volatile("global_load_async_to_lds_b128 %0, %1, off"
               :: "v"(lds_off), "v"((unsigned long long)(uintptr_t)g)
               : "memory");
}
template <int N>
__device__ __forceinline__ void wait_async() {
  asm volatile("s_wait_asynccnt %0" :: "i"(N) : "memory");
}

// ---------------------------------------------------------------------------
// Kernel 0: one-shot f32 -> f16 conversion (memory bound).
// Keeps all v_cvt out of the GEMM hot loops.
// ---------------------------------------------------------------------------
__global__ __launch_bounds__(256) void cvt_f16_kernel(
    const float* __restrict__ src, _Float16* __restrict__ dst, int n8) {
  const int i = blockIdx.x * blockDim.x + threadIdx.x;
  if (i >= n8) return;
  union { v8h v; _Float16 e[8]; } u;
  const float* p = src + (size_t)i * 8;
#pragma unroll
  for (int j = 0; j < 8; ++j) u.e[j] = (_Float16)p[j];
  *(v8h*)(dst + (size_t)i * 8) = u.v;
}

// ---------------------------------------------------------------------------
// Kernel 1: fused QKV projection + RoPE, software-pipelined weight fragments.
// Branch-free inner loop; last E-chunk peeled into an epilogue.
// One wave handles 16 sequence rows; contraction over E=1024 in K=32 chunks.
// Outputs: Qh,Kh f16 (B*S, 64) row-major; V stored transposed (B, 64, S).
// ---------------------------------------------------------------------------
__global__ __launch_bounds__(32) void qkv_rope_kernel(
    const _Float16* __restrict__ Xh, const _Float16* __restrict__ Wqh,
    const _Float16* __restrict__ Wkh, const _Float16* __restrict__ Wvh,
    _Float16* __restrict__ Qh, _Float16* __restrict__ Kh,
    _Float16* __restrict__ Vt) {
  const int lane = threadIdx.x & 31;
  const int lh   = lane >> 4;        // wave half
  const int nl   = lane & 15;        // row (A) / col (B) within tile
  const int ch   = lh * 8;           // K-offset base for this lane
  const int g0   = blockIdx.x * 16;  // first flattened (b*S+s) row

  v8f accq[4], acck[4], accv[4];
#pragma unroll
  for (int t = 0; t < 4; ++t) { accq[t] = zero8(); acck[t] = zero8(); accv[t] = zero8(); }

  const _Float16* xrow = Xh + (size_t)(g0 + nl) * E_DIM;

  v16h fq[4], fk[4], fv[4];
  auto loadW4 = [&](const _Float16* __restrict__ W, int ec, v16h* dst) {
#pragma unroll
    for (int t = 0; t < 4; ++t)
      dst[t] = frag_from_f16(W + (size_t)(t * 16 + nl) * E_DIM, ec, ch);
  };
  auto wmma4 = [&](const v16h& a, const v16h* bf, v8f* acc) {
#pragma unroll
    for (int t = 0; t < 4; ++t)
      acc[t] = __builtin_amdgcn_wmma_f32_16x16x32_f16(
          false, a, false, bf[t], (short)0, acc[t], false, false);
  };

  // Prologue: first-chunk Q/K weight fragments + first A fragment in flight.
  loadW4(Wqh, 0, fq);
  loadW4(Wkh, 0, fk);
  v16h a = frag_from_f16(xrow, 0, ch);

  // Main loop: all next-chunk loads unconditional (last chunk peeled below).
  int ec = 0;
  for (; ec < E_DIM - 32; ec += 32) {
    __builtin_prefetch(xrow + ec + 64, 0, 1);  // global_prefetch_b8 (speculative)
    loadW4(Wvh, ec, fv);
    const v16h acur = a;
    a = frag_from_f16(xrow, ec + 32, ch);      // next A while Q-WMMAs run
    wmma4(acur, fq, accq);
    loadW4(Wqh, ec + 32, fq);                  // next-chunk Q while K-WMMAs run
    wmma4(acur, fk, acck);
    loadW4(Wkh, ec + 32, fk);                  // next-chunk K while V-WMMAs run
    wmma4(acur, fv, accv);
  }
  // Epilogue chunk (ec == E_DIM-32): no further loads.
  loadW4(Wvh, ec, fv);
  wmma4(a, fq, accq);
  wmma4(a, fk, acck);
  wmma4(a, fv, accv);

  // RoPE in C/D layout: lane l holds (M = r + 8*(l>=16), N = l%16).
  // Feature pair (h, h^1) lives in adjacent lanes -> shfl_xor(.,1).
#pragma unroll
  for (int t = 0; t < 4; ++t) {
    const int h = t * 16 + nl;
    const int p = h >> 1;
    const float fr = __expf(-(float)(2 * p) * (9.210340371976184f / 64.f));
#pragma unroll
    for (int r = 0; r < 8; ++r) {
      const int grow = g0 + r + 8 * lh;
      const int spos = grow & (S_LEN - 1);
      const float ang = (float)spos * fr;
      const float cs = __cosf(ang), sn = __sinf(ang);

      float vq = accq[t][r];
      float oq = __shfl_xor(vq, 1, 32);
      float nq = (h & 1) ? (oq * sn + vq * cs) : (vq * cs - oq * sn);
      Qh[(size_t)grow * H_DIM + h] = (_Float16)nq;

      float vk = acck[t][r];
      float ok = __shfl_xor(vk, 1, 32);
      float nk = (h & 1) ? (ok * sn + vk * cs) : (vk * cs - ok * sn);
      Kh[(size_t)grow * H_DIM + h] = (_Float16)nk;

      const int b = grow >> 12;  // /S_LEN
      Vt[((size_t)(b * H_DIM + h)) * S_LEN + spos] = (_Float16)accv[t][r];
    }
  }
}

// ---------------------------------------------------------------------------
// Kernel 2: causal flash attention, one wave per 16-query tile.
// K/V chunks double-buffered in LDS via global_load_async_to_lds_b128
// (ASYNCcnt), fragments pulled from LDS with ds_load_b128.
// Per 32-key chunk: 4 WMMAs (scores) + online softmax + LDS transpose of P
// (C/D layout -> A layout) + 4 WMMAs (P@V).
// ---------------------------------------------------------------------------
__global__ __launch_bounds__(32) void flash_attn_kernel(
    const _Float16* __restrict__ Qh, const _Float16* __restrict__ Kh,
    const _Float16* __restrict__ Vt, const unsigned char* __restrict__ pmask,
    float* __restrict__ out) {
  __shared__ __align__(16) _Float16 Klds[2][32 * 64];  // key-row x h, 4KB each
  __shared__ __align__(16) _Float16 Vlds[2][64 * 32];  // h-row x key, 4KB each
  __shared__ __align__(16) _Float16 Plds[16 * 48];     // P transpose staging

  const int lane = threadIdx.x & 31;
  const int lh   = lane >> 4;
  const int nl   = lane & 15;
  const int ch   = lh * 8;
  const int b    = blockIdx.y;
  const int q0   = blockIdx.x * 16;
  const size_t base = (size_t)b * S_LEN;

  const _Float16* kbase_g = Kh + base * H_DIM;
  const _Float16* vbase_g = Vt + (size_t)b * H_DIM * S_LEN;

  // Stage one 32-key chunk of K (4KB, linear) and V (64 h-rows x 64B) to LDS.
  // 16 async b128 ops per chunk; all lanes participate.
  auto stage = [&](int bufi, int kc) {
    const char* kg = (const char*)(kbase_g + (size_t)kc * H_DIM);
    const unsigned kl = (unsigned)(uintptr_t)&Klds[bufi][0];
#pragma unroll
    for (int i = 0; i < 8; ++i) {
      const int task = i * 32 + lane;          // 256 x 16B = 4KB linear
      async_load_b128(kl + task * 16, kg + task * 16);
    }
    const unsigned vl = (unsigned)(uintptr_t)&Vlds[bufi][0];
#pragma unroll
    for (int i = 0; i < 8; ++i) {
      const int task = i * 32 + lane;          // h = task/4, 16B segment
      const int h = task >> 2, seg = task & 3;
      async_load_b128(vl + task * 16,
                      (const char*)(vbase_g + (size_t)h * S_LEN + kc) + seg * 16);
    }
  };

  // Persistent Q fragments: 16 queries x 64 features (two K=32 chunks)
  const _Float16* qrow = Qh + (base + q0 + nl) * H_DIM;
  const v16h aq0 = frag_from_f16(qrow, 0, ch);
  const v16h aq1 = frag_from_f16(qrow, 32, ch);

  v8f accO[4];
#pragma unroll
  for (int t = 0; t < 4; ++t) accO[t] = zero8();
  float mrow[8], lrow[8];
#pragma unroll
  for (int r = 0; r < 8; ++r) { mrow[r] = -__builtin_inff(); lrow[r] = 0.f; }

  const float scale = 0.125f;  // 1/sqrt(64)
  const int kend = q0 + 16;    // causal: keys 0..q0+15

  stage(0, 0);
  int buf = 0;

  for (int kc = 0; kc < kend; kc += 32) {
    const bool more = (kc + 32 < kend);
    if (more) stage(buf ^ 1, kc + 32);   // prefetch next chunk into other buffer
    if (more) wait_async<16>();          // in-order: current chunk has landed
    else      wait_async<0>();

    const _Float16* Kb = &Klds[buf][0];
    const _Float16* Vb = &Vlds[buf][0];

    // Scores: two 16(q) x 16(key) tiles, contraction over H=64 (from LDS).
    v8f sc[2];
#pragma unroll
    for (int t = 0; t < 2; ++t) {
      const _Float16* krow = Kb + (t * 16 + nl) * H_DIM;  // B col n = key
      v16h b0 = frag_from_f16(krow, 0, ch);
      v16h b1 = frag_from_f16(krow, 32, ch);
      v8f z = zero8();
      z = __builtin_amdgcn_wmma_f32_16x16x32_f16(false, aq0, false, b0,
                                                 (short)0, z, false, false);
      z = __builtin_amdgcn_wmma_f32_16x16x32_f16(false, aq1, false, b1,
                                                 (short)0, z, false, false);
      sc[t] = z;
    }

    // Online softmax per query row (row = (r, wave-half); N spread over lanes)
    float pv[2][8];
#pragma unroll
    for (int r = 0; r < 8; ++r) {
      const int qg = q0 + r + 8 * lh;
      float m0 = -__builtin_inff();
#pragma unroll
      for (int t = 0; t < 2; ++t) {
        const int kg = kc + t * 16 + nl;
        float v = sc[t][r] * scale;
        if (kg > qg) v = -__builtin_inff();  // causal mask
        pv[t][r] = v;
        m0 = fmaxf(m0, v);
      }
#pragma unroll
      for (int off = 1; off < 16; off <<= 1)
        m0 = fmaxf(m0, __shfl_xor(m0, off, 32));
      const float mn = fmaxf(mrow[r], m0);
      const float alpha = __expf(mrow[r] - mn);  // 0 on first chunk
      float s0 = 0.f;
#pragma unroll
      for (int t = 0; t < 2; ++t) {
        const float e = __expf(pv[t][r] - mn);
        pv[t][r] = e;
        s0 += e;
      }
#pragma unroll
      for (int off = 1; off < 16; off <<= 1) s0 += __shfl_xor(s0, off, 32);
      lrow[r] = lrow[r] * alpha + s0;  // masked keys stay in denominator
      mrow[r] = mn;
#pragma unroll
      for (int t = 0; t < 4; ++t) accO[t][r] *= alpha;
      // Reference zeroes padded keys AFTER softmax -> zero P but keep lrow.
#pragma unroll
      for (int t = 0; t < 2; ++t) {
        const int kg = kc + t * 16 + nl;
        if (pmask[base + kg]) pv[t][r] = 0.f;
        Plds[(r + 8 * lh) * 48 + t * 16 + nl] = (_Float16)pv[t][r];
      }
    }
    __syncthreads();

    // Re-read P in A-fragment layout (row m = nl, padded stride 48 halves)
    v16h pa;
    {
      union { v16h v; struct { v8h lo, hi; } s; } u;
      const _Float16* prow = &Plds[nl * 48];
      u.s.lo = *(const v8h*)(prow + ch);
      u.s.hi = *(const v8h*)(prow + ch + 16);
      pa = u.v;
    }

    // P (16x32) @ V (32 keys x 64 h): V rows in LDS give contiguous keys.
#pragma unroll
    for (int t = 0; t < 4; ++t) {
      const _Float16* vrow = Vb + (t * 16 + nl) * 32;  // B col n = h
      v16h bv = frag_from_f16(vrow, 0, ch);
      accO[t] = __builtin_amdgcn_wmma_f32_16x16x32_f16(
          false, pa, false, bv, (short)0, accO[t], false, false);
    }
    __syncthreads();
    buf ^= 1;
  }

  // Final normalize + store (fp32 output)
#pragma unroll
  for (int t = 0; t < 4; ++t)
#pragma unroll
    for (int r = 0; r < 8; ++r) {
      const int qg = q0 + r + 8 * lh;
      out[(base + qg) * H_DIM + t * 16 + nl] = accO[t][r] / lrow[r];
    }
}

extern "C" void kernel_launch(void* const* d_in, const int* in_sizes, int n_in,
                              void* d_out, int out_size, void* d_ws, size_t ws_size,
                              hipStream_t stream) {
  const float* x  = (const float*)d_in[0];
  const unsigned char* pm = (const unsigned char*)d_in[1];  // bool mask (B,1,S)
  const float* Wq = (const float*)d_in[2];
  const float* Wk = (const float*)d_in[3];
  const float* Wv = (const float*)d_in[4];
  float* out = (float*)d_out;

  const size_t nBS = (size_t)4 * S_LEN;  // 16384 rows
  _Float16* Qh  = (_Float16*)d_ws;              // 2 MB
  _Float16* Kh  = Qh + nBS * H_DIM;             // 2 MB
  _Float16* Vt  = Kh + nBS * H_DIM;             // 2 MB (B, 64, S)
  _Float16* Xh  = Vt + nBS * H_DIM;             // 32 MB (B*S, E) f16
  _Float16* Wqh = Xh + nBS * E_DIM;             // 128 KB each
  _Float16* Wkh = Wqh + (size_t)H_DIM * E_DIM;
  _Float16* Wvh = Wkh + (size_t)H_DIM * E_DIM;
  (void)in_sizes; (void)n_in; (void)out_size; (void)ws_size;

  // One-shot f32->f16 conversions (keeps v_cvt out of the GEMM loops).
  {
    const int nx8 = (int)(nBS * E_DIM / 8);
    cvt_f16_kernel<<<dim3((nx8 + 255) / 256), 256, 0, stream>>>(x, Xh, nx8);
    const int nw8 = H_DIM * E_DIM / 8;
    cvt_f16_kernel<<<dim3((nw8 + 255) / 256), 256, 0, stream>>>(Wq, Wqh, nw8);
    cvt_f16_kernel<<<dim3((nw8 + 255) / 256), 256, 0, stream>>>(Wk, Wkh, nw8);
    cvt_f16_kernel<<<dim3((nw8 + 255) / 256), 256, 0, stream>>>(Wv, Wvh, nw8);
  }

  qkv_rope_kernel<<<dim3(nBS / 16), 32, 0, stream>>>(Xh, Wqh, Wkh, Wvh, Qh, Kh, Vt);
  flash_attn_kernel<<<dim3(S_LEN / 16, 4), 32, 0, stream>>>(Qh, Kh, Vt, pm, out);
}
